// PerceiverAttention_89378269430421
// MI455X (gfx1250) — matrix operand order repaired and checked by reading, not compiled, not run on hardware
//
#include <hip/hip_runtime.h>

// ---------------- types ----------------
typedef __bf16 bf16_t;
typedef __attribute__((ext_vector_type(16))) __bf16 v16bf;
typedef __attribute__((ext_vector_type(8)))  __bf16 v8bf;
typedef __attribute__((ext_vector_type(4)))  __bf16 v4bf;
typedef __attribute__((ext_vector_type(8)))  float  v8f;
typedef __attribute__((vector_size(16)))     int    vint4;

// ---------------- problem constants ----------------
constexpr int kDIM   = 1024;
constexpr int kHEADS = 16;
constexpr int kHDIM  = 64;
constexpr int kL     = 512;
constexpr int kS     = 8192;
constexpr int kB     = 4;
// SCALE * log2(e) = 64^-0.5 * 1.4426950408889634
constexpr float kQSCALE = 0.18033688f;

__device__ __forceinline__ v16bf cat8(v8bf lo, v8bf hi) {
  return __builtin_shufflevector(lo, hi, 0,1,2,3,4,5,6,7,8,9,10,11,12,13,14,15);
}

__device__ __forceinline__ v8f wmma_bf16(v16bf a, v16bf b, v8f c) {
  return __builtin_amdgcn_wmma_f32_16x16x32_bf16(false, a, false, b, (short)0, c,
                                                 false, false);
}

// ---------------- CDNA5 async copy global->LDS (ASYNCcnt path) --------------
#if __has_builtin(__builtin_amdgcn_global_load_async_to_lds_b128)
#define HAVE_ASYNC_LDS 1
__device__ __forceinline__ void async_cp16(const bf16_t* g, bf16_t* l) {
  __builtin_amdgcn_global_load_async_to_lds_b128((vint4*)g, (vint4*)l, 0, 0);
}
#else
#define HAVE_ASYNC_LDS 0
__device__ __forceinline__ void async_cp16(const bf16_t* g, bf16_t* l) {
  *(v8bf*)l = *(const v8bf*)g;  // fallback: synchronous copy through VGPRs
}
#endif

#if HAVE_ASYNC_LDS
#if __has_builtin(__builtin_amdgcn_s_wait_asynccnt)
#define WAIT_ASYNC(n) __builtin_amdgcn_s_wait_asynccnt(n)
#else
#define WAIT_ASYNC(n) asm volatile("s_wait_asynccnt %0" ::"i"(n) : "memory")
#endif
#else
#define WAIT_ASYNC(n) ((void)0)
#endif

// wave-level LDS write->read fence (LDS ops of a wave are in-order; this just
// forces the dscnt wait and stops compiler reordering)
__device__ __forceinline__ void lds_fence_wave() {
  asm volatile("s_wait_dscnt 0x0" ::: "memory");
}

// ---------------- f32 -> bf16 conversion ----------------
__global__ __launch_bounds__(256) void cvt_f32_bf16(const float* __restrict__ in,
                                                    bf16_t* __restrict__ out,
                                                    long n) {
  long i = ((long)blockIdx.x * blockDim.x + threadIdx.x) * 4;
  if (i + 3 < n) {
    float4 f = *(const float4*)(in + i);
    v4bf o;
    o.x = (bf16_t)f.x; o.y = (bf16_t)f.y; o.z = (bf16_t)f.z; o.w = (bf16_t)f.w;
    *(v4bf*)(out + i) = o;
  }
}

// ---------------- shared WMMA GEMM core (LDS triple-buffered) ----------------
// Block tile 128(M) x 128(N), K-step 64. 8 waves as 4(M) x 2(N): each wave owns
// 32x64 = 2x4 WMMA accumulators (16 WMMAs per K-step). A/W tiles staged in LDS
// via async copies, 3 buffers, 2 tiles in flight, ONE barrier per K-step.
__device__ __forceinline__ void gemm_core_lds(const bf16_t* __restrict__ A,
                                              const bf16_t* __restrict__ W,
                                              int K, long mblk, long nblk,
                                              bf16_t* a_lds, bf16_t* w_lds,
                                              v8f acc[2][4]) {
  const int t = threadIdx.x;
  const int lane = t & 31, hl = lane >> 4, lr = lane & 15;
  const int wave = t >> 5, mw = wave >> 1, nw = wave & 1;
  const int nsteps = K / 64;

  auto issue = [&](int kt, int buf) {
    const long k0 = (long)kt * 64;
    bf16_t* al = a_lds + buf * (128 * 64);
    bf16_t* wl = w_lds + buf * (128 * 64);
#pragma unroll
    for (int c = 0; c < 4; ++c) {
      const int chunk = t + c * 256;           // 1024 16B-chunks per 16KB tile
      const int row = chunk >> 3, sub = (chunk & 7) * 8;
      async_cp16(A + (mblk + row) * (long)K + k0 + sub, al + row * 64 + sub);
      async_cp16(W + (nblk + row) * (long)K + k0 + sub, wl + row * 64 + sub);
    }
  };

  issue(0, 0);
  if (nsteps > 1) issue(1, 1);
  for (int kt = 0; kt < nsteps; ++kt) {
    const int cur = kt % 3;
    // own tile-kt copies done (tile kt+1 may stay in flight)
    if (kt + 1 < nsteps) WAIT_ASYNC(8); else WAIT_ASYNC(0);
    __syncthreads();   // all waves' tile-kt copies visible; all done reading kt-1
    if (kt + 2 < nsteps) issue(kt + 2, (kt + 2) % 3);

    const bf16_t* al = a_lds + cur * (128 * 64);
    const bf16_t* wl = w_lds + cur * (128 * 64);
#pragma unroll
    for (int kh = 0; kh < 2; ++kh) {
      v16bf af[2], bfb[4];
#pragma unroll
      for (int mt = 0; mt < 2; ++mt) {
        const bf16_t* ar = al + (mw * 32 + mt * 16 + lr) * 64 + kh * 32;
        af[mt] = *(const v16bf*)(ar + hl * 16);
      }
#pragma unroll
      for (int nb = 0; nb < 4; ++nb)
        bfb[nb] = *(const v16bf*)(wl + (nw * 64 + nb * 16 + lr) * 64 + kh * 32 + hl * 16);
#pragma unroll
      for (int mt = 0; mt < 2; ++mt)
#pragma unroll
        for (int nb = 0; nb < 4; ++nb)
          acc[mt][nb] = wmma_bf16(af[mt], bfb[nb], acc[mt][nb]);
    }
  }
}

// ---------------- q projection ------------------------------------------------
__global__ __launch_bounds__(256) void gemm_q_kernel(const bf16_t* __restrict__ A,
                                                     const bf16_t* __restrict__ W,
                                                     bf16_t* __restrict__ C) {
  __shared__ __align__(16) bf16_t a_lds[3 * 128 * 64];
  __shared__ __align__(16) bf16_t w_lds[3 * 128 * 64];
  const long mblk = (long)blockIdx.y * 128, nblk = (long)blockIdx.x * 128;
  v8f acc[2][4] = {};
  gemm_core_lds(A, W, kDIM, mblk, nblk, a_lds, w_lds, acc);
  const int lane = threadIdx.x & 31, hl = lane >> 4, lr = lane & 15;
  const int wave = threadIdx.x >> 5, mw = wave >> 1, nw = wave & 1;
#pragma unroll
  for (int mt = 0; mt < 2; ++mt)
#pragma unroll
    for (int nb = 0; nb < 4; ++nb)
#pragma unroll
      for (int i = 0; i < 8; ++i)
        C[(mblk + mw * 32 + mt * 16 + i + 8 * hl) * (long)kDIM +
          nblk + nw * 64 + nb * 16 + lr] = (bf16_t)(acc[mt][nb][i] * kQSCALE);
}

// ---------------- kv projection: k row-major, v transposed (B,H,64,S) --------
__global__ __launch_bounds__(256) void gemm_kv_kernel(const bf16_t* __restrict__ X,
                                                      const bf16_t* __restrict__ Wkv,
                                                      bf16_t* __restrict__ Kout,
                                                      bf16_t* __restrict__ VTout) {
  __shared__ __align__(16) bf16_t a_lds[3 * 128 * 64];
  __shared__ __align__(16) bf16_t w_lds[3 * 128 * 64];
  const long mblk = (long)blockIdx.y * 128, nblk = (long)blockIdx.x * 128;
  v8f acc[2][4] = {};
  gemm_core_lds(X, Wkv, kDIM, mblk, nblk, a_lds, w_lds, acc);
  const int lane = threadIdx.x & 31, hl = lane >> 4, lr = lane & 15;
  const int wave = threadIdx.x >> 5, mw = wave >> 1, nw = wave & 1;
#pragma unroll
  for (int mt = 0; mt < 2; ++mt)
#pragma unroll
    for (int nb = 0; nb < 4; ++nb) {
      const long n = nblk + nw * 64 + nb * 16 + lr;
#pragma unroll
      for (int i = 0; i < 8; ++i) {
        const long row = mblk + mw * 32 + mt * 16 + i + 8 * hl;  // b*S + s
        const float vv = acc[mt][nb][i];
        if (n < kDIM) {
          Kout[row * (long)kDIM + n] = (bf16_t)vv;
        } else {
          const long d = n - kDIM;
          const long b = row >> 13, s = row & (kS - 1);
          const long h = d >> 6, dd = d & 63;
          VTout[(((b * kHEADS + h) * kHDIM) + dd) * (long)kS + s] = (bf16_t)vv;
        }
      }
    }
}

// ---------------- output projection: f32 out + bias -------------------------
__global__ __launch_bounds__(256) void gemm_out_kernel(const bf16_t* __restrict__ A,
                                                       const bf16_t* __restrict__ W,
                                                       const float* __restrict__ bias,
                                                       float* __restrict__ Out) {
  __shared__ __align__(16) bf16_t a_lds[3 * 128 * 64];
  __shared__ __align__(16) bf16_t w_lds[3 * 128 * 64];
  const long mblk = (long)blockIdx.y * 128, nblk = (long)blockIdx.x * 128;
  v8f acc[2][4] = {};
  gemm_core_lds(A, W, kDIM, mblk, nblk, a_lds, w_lds, acc);
  const int lane = threadIdx.x & 31, hl = lane >> 4, lr = lane & 15;
  const int wave = threadIdx.x >> 5, mw = wave >> 1, nw = wave & 1;
#pragma unroll
  for (int mt = 0; mt < 2; ++mt)
#pragma unroll
    for (int nb = 0; nb < 4; ++nb) {
      const float bv = bias[nblk + nw * 64 + nb * 16 + lr];
#pragma unroll
      for (int i = 0; i < 8; ++i)
        Out[(mblk + mw * 32 + mt * 16 + i + 8 * hl) * (long)kDIM +
            nblk + nw * 64 + nb * 16 + lr] = acc[mt][nb][i] + bv;
    }
}

// ---------------- flash attention --------------------------------------------
// grid: (B*H, L/128); block 256 = 8 waves; wave owns 16 q rows. 64-wide S
// chunks of K/V staged in LDS (async, triple-buffered, one barrier per chunk).
// 16 WMMAs per chunk per wave between barriers.
__global__ __launch_bounds__(256) void attn_kernel(const bf16_t* __restrict__ Q,
                                                   const bf16_t* __restrict__ Kc,
                                                   const bf16_t* __restrict__ VT,
                                                   bf16_t* __restrict__ Ao) {
  __shared__ __align__(16) bf16_t k_lds[3][64 * 64];  // [s][d]
  __shared__ __align__(16) bf16_t v_lds[3][64 * 64];  // [d][s]
  __shared__ __align__(16) bf16_t plds[8][16 * 64];   // per-wave P staging
  const int t = threadIdx.x;
  const int wave = t >> 5, lane = t & 31;
  const int hl = lane >> 4, lr = lane & 15;
  const int bh = blockIdx.x;
  const int b = bh >> 4, h = bh & 15;
  const long l0 = (long)blockIdx.y * 128 + wave * 16;

  // Q fragments kept in registers for the whole S loop
  const bf16_t* qrow = Q + ((long)b * kL + l0 + lr) * (long)kDIM + h * kHDIM;
  const v16bf qa0 = *(const v16bf*)(qrow + hl * 16);
  const v16bf qa1 = *(const v16bf*)(qrow + 32 + hl * 16);

  v8f o[4] = {};
  float mrow[8], lrow[8];
#pragma unroll
  for (int i = 0; i < 8; ++i) { mrow[i] = -3.0e38f; lrow[i] = 0.f; }

  const bf16_t* kgbase = Kc + ((long)b * kS) * (long)kDIM + h * kHDIM;
  const bf16_t* vgbase = VT + ((long)bh * kHDIM) * (long)kS;
  bf16_t* pw = plds[wave];

  auto issue = [&](long s0, int buf) {
#pragma unroll
    for (int c = 0; c < 2; ++c) {
      const int chunk = t + c * 256;           // 512 16B-chunks per 8KB tile
      const int row = chunk >> 3, sub = (chunk & 7) * 8;
      // K chunk: 64 s-rows x 64 d
      async_cp16(kgbase + (s0 + row) * (long)kDIM + sub,
                 &k_lds[buf][row * 64 + sub]);
      // V chunk (transposed): 64 d-rows x 64 s
      async_cp16(vgbase + (long)row * kS + s0 + sub, &v_lds[buf][row * 64 + sub]);
    }
  };

  const int nsteps = kS / 64;
  issue(0, 0);
  issue(64, 1);
  for (int it = 0; it < nsteps; ++it) {
    const int cur = it % 3;
    if (it + 1 < nsteps) WAIT_ASYNC(4); else WAIT_ASYNC(0);
    __syncthreads();
    if (it + 2 < nsteps) issue((long)(it + 2) * 64, (it + 2) % 3);

    // ---- scores 16x64 via 8 WMMAs; K fragments from LDS ----
    const bf16_t* kl = k_lds[cur];
    v8f sc[4];
#pragma unroll
    for (int ct = 0; ct < 4; ++ct) {
      const bf16_t* kr = kl + (ct * 16 + lr) * 64 + hl * 16;
      v16bf klo = *(const v16bf*)(kr);
      v16bf khi = *(const v16bf*)(kr + 32);
      v8f s = {};
      s = wmma_bf16(qa0, klo, s);
      s = wmma_bf16(qa1, khi, s);
      sc[ct] = s;
    }

    // ---- online softmax (rows live across 16-lane groups) ----
    float nm[8];
#pragma unroll
    for (int i = 0; i < 8; ++i)
      nm[i] = fmaxf(fmaxf(sc[0][i], sc[1][i]), fmaxf(sc[2][i], sc[3][i]));
#pragma unroll
    for (int msk = 1; msk < 16; msk <<= 1)
#pragma unroll
      for (int i = 0; i < 8; ++i) nm[i] = fmaxf(nm[i], __shfl_xor(nm[i], msk, 16));
    float alpha[8];
#pragma unroll
    for (int i = 0; i < 8; ++i) {
      const float newm = fmaxf(mrow[i], nm[i]);
      alpha[i] = exp2f(mrow[i] - newm);
      mrow[i] = newm;
      lrow[i] *= alpha[i];
    }
#pragma unroll
    for (int nb = 0; nb < 4; ++nb)
#pragma unroll
      for (int i = 0; i < 8; ++i) o[nb][i] *= alpha[i];

    // ---- P = exp2(S - m): C layout -> LDS row-major 16x64 ----
#pragma unroll
    for (int ct = 0; ct < 4; ++ct)
#pragma unroll
      for (int i = 0; i < 8; ++i) {
        const float p = exp2f(sc[ct][i] - mrow[i]);
        lrow[i] += p;
        pw[(i + 8 * hl) * 64 + ct * 16 + lr] = (bf16_t)p;
      }
    lds_fence_wave();   // wave-private region: dscnt fence is enough

    // ---- P back as two A fragments (K = 64) ----
    const bf16_t* pr = pw + lr * 64;
    const v16bf pa0 = *(const v16bf*)(pr + hl * 16);
    const v16bf pa1 = *(const v16bf*)(pr + 32 + hl * 16);

    // ---- O += P(16x64) x V(64x64) via 8 WMMAs ----
    const bf16_t* vl = v_lds[cur];
#pragma unroll
    for (int nb = 0; nb < 4; ++nb) {
      const bf16_t* vr = vl + (nb * 16 + lr) * 64 + hl * 16;
      v16bf vb0 = *(const v16bf*)(vr);
      v16bf vb1 = *(const v16bf*)(vr + 32);
      o[nb] = wmma_bf16(pa0, vb0, o[nb]);
      o[nb] = wmma_bf16(pa1, vb1, o[nb]);
    }
  }

  // ---- normalize and store bf16 (B,L,DIM) ----
#pragma unroll
  for (int msk = 1; msk < 16; msk <<= 1)
#pragma unroll
    for (int i = 0; i < 8; ++i) lrow[i] += __shfl_xor(lrow[i], msk, 16);
#pragma unroll
  for (int i = 0; i < 8; ++i) lrow[i] = 1.0f / lrow[i];
#pragma unroll
  for (int nb = 0; nb < 4; ++nb)
#pragma unroll
    for (int i = 0; i < 8; ++i) {
      const float val = o[nb][i] * lrow[i];
      Ao[((long)b * kL + l0 + i + 8 * hl) * (long)kDIM + h * kHDIM + nb * 16 + lr] =
          (bf16_t)val;
    }
}

// ---------------- host launcher ----------------------------------------------
extern "C" void kernel_launch(void* const* d_in, const int* in_sizes, int n_in,
                              void* d_out, int out_size, void* d_ws, size_t ws_size,
                              hipStream_t stream) {
  const float* x    = (const float*)d_in[0];
  const float* lat  = (const float*)d_in[1];
  const float* wq   = (const float*)d_in[2];
  const float* wkv  = (const float*)d_in[3];
  const float* wout = (const float*)d_in[4];
  const float* bout = (const float*)d_in[5];
  float* out = (float*)d_out;

  char* ws = (char*)d_ws;
  size_t off = 0;
  auto alloc = [&](size_t bytes) -> char* {
    char* p = ws + off;
    off += (bytes + 255) & ~(size_t)255;
    return p;
  };
  bf16_t* xb    = (bf16_t*)alloc((size_t)kB * kS * kDIM * 2);
  bf16_t* latb  = (bf16_t*)alloc((size_t)kB * kL * kDIM * 2);
  bf16_t* wqb   = (bf16_t*)alloc((size_t)kDIM * kDIM * 2);
  bf16_t* wkvb  = (bf16_t*)alloc((size_t)2 * kDIM * kDIM * 2);
  bf16_t* woutb = (bf16_t*)alloc((size_t)kDIM * kDIM * 2);
  bf16_t* qb    = (bf16_t*)alloc((size_t)kB * kL * kDIM * 2);
  bf16_t* kb    = (bf16_t*)alloc((size_t)kB * kS * kDIM * 2);
  bf16_t* vtb   = (bf16_t*)alloc((size_t)kB * kS * kDIM * 2);
  bf16_t* aob   = (bf16_t*)alloc((size_t)kB * kL * kDIM * 2);

  auto cvt = [&](const float* src, bf16_t* dst, long n) {
    long thr = n / 4;
    cvt_f32_bf16<<<dim3((unsigned)((thr + 255) / 256)), dim3(256), 0, stream>>>(src, dst, n);
  };
  cvt(x,    xb,    (long)kB * kS * kDIM);
  cvt(lat,  latb,  (long)kB * kL * kDIM);
  cvt(wq,   wqb,   (long)kDIM * kDIM);
  cvt(wkv,  wkvb,  (long)2 * kDIM * kDIM);
  cvt(wout, woutb, (long)kDIM * kDIM);

  gemm_q_kernel<<<dim3(kDIM / 128, (kB * kL) / 128), 256, 0, stream>>>(latb, wqb, qb);
  gemm_kv_kernel<<<dim3((2 * kDIM) / 128, (kB * kS) / 128), 256, 0, stream>>>(xb, wkvb, kb, vtb);
  attn_kernel<<<dim3(kB * kHEADS, kL / 128), 256, 0, stream>>>(qb, kb, vtb, aob);
  gemm_out_kernel<<<dim3(kDIM / 128, (kB * kL) / 128), 256, 0, stream>>>(aob, woutb, bout, out);
}
